// MaskedCausalAttention_17282948399783
// MI455X (gfx1250) — matrix operand order repaired and checked by reading, not compiled, not run on hardware
//
#include <hip/hip_runtime.h>
#include <stdint.h>

// Problem constants (reference: B=2, T=2048, C=1024, 16 heads, D=64)
#define BB   2
#define TT   2048
#define CC   1024
#define NH   16
#define HD   64
#define MTOT (BB * TT)   // 4096 rows

typedef __attribute__((ext_vector_type(16))) __bf16 v16bf;
typedef __attribute__((ext_vector_type(8)))  float  v8f;
typedef __attribute__((ext_vector_type(4)))  float  f32x4;
typedef __attribute__((ext_vector_type(4)))  int    i32x4;

// 32-byte WMMA operand fragment (8 VGPRs).
union FragB16 {
    f32x4    f4[2];
    uint32_t u[8];
    v16bf    v;
};

#define LD4(p) (*reinterpret_cast<const f32x4*>(p))

// ---------------- async global->LDS copy (gfx1250 ASYNCcnt path) -----------
#if __has_builtin(__builtin_amdgcn_global_load_async_to_lds_b128)
#define HAVE_ASYNC 1
#else
#define HAVE_ASYNC 0
#endif

__device__ __forceinline__ void async_cp16(uint16_t* lds_dst, const uint16_t* gsrc) {
#if HAVE_ASYNC
    __builtin_amdgcn_global_load_async_to_lds_b128(
        (__attribute__((address_space(1))) i32x4*)gsrc,
        (__attribute__((address_space(3))) i32x4*)lds_dst,
        0, 0);
#else
    *reinterpret_cast<f32x4*>(lds_dst) = LD4(gsrc);   // sync fallback
#endif
}

#if HAVE_ASYNC
#  if __has_builtin(__builtin_amdgcn_s_wait_asynccnt)
#    define WAIT_ASYNCCNT(n) __builtin_amdgcn_s_wait_asynccnt(n)
#  else
#    define WAIT_ASYNCCNT(n) asm volatile("s_wait_asynccnt %0" ::"n"(n) : "memory")
#  endif
#else
#  define WAIT_ASYNCCNT(n) ((void)0)
#endif

// ---------------- DPP 16-lane row reductions (pure VALU, no LDS) -----------
// quad_perm xor1 = 0xB1, quad_perm xor2 = 0x4E, row_ror:4 = 0x124, row_ror:8 = 0x128
template <int CTRL>
__device__ __forceinline__ float dpp_mov_f32(float x) {
    int r = __builtin_amdgcn_update_dpp(0, __builtin_bit_cast(int, x),
                                        CTRL, 0xF, 0xF, true);
    return __builtin_bit_cast(float, r);
}
__device__ __forceinline__ float row16_max(float v) {
    v = fmaxf(v, dpp_mov_f32<0xB1>(v));
    v = fmaxf(v, dpp_mov_f32<0x4E>(v));
    v = fmaxf(v, dpp_mov_f32<0x124>(v));
    v = fmaxf(v, dpp_mov_f32<0x128>(v));
    return v;
}
__device__ __forceinline__ float row16_sum(float v) {
    v += dpp_mov_f32<0xB1>(v);
    v += dpp_mov_f32<0x4E>(v);
    v += dpp_mov_f32<0x124>(v);
    v += dpp_mov_f32<0x128>(v);
    return v;
}

__device__ __forceinline__ uint16_t f2bf(float x) {
    union { float f; uint32_t u; } c; c.f = x;
    uint32_t r = c.u + 0x7FFFu + ((c.u >> 16) & 1u);  // round-to-nearest-even
    return (uint16_t)(r >> 16);
}

__device__ __forceinline__ v8f wmma_bf16(const FragB16& a, const FragB16& b, v8f c) {
    return __builtin_amdgcn_wmma_f32_16x16x32_bf16(false, a.v, false, b.v,
                                                   (short)0, c, false, false);
}

// ---------------------------------------------------------------------------
// fp32 -> bf16 elementwise convert
// ---------------------------------------------------------------------------
__global__ void cvt_f32_bf16(const float* __restrict__ in,
                             uint16_t* __restrict__ out, int n) {
    int i = blockIdx.x * blockDim.x + threadIdx.x;
    if (i < n) out[i] = f2bf(in[i]);
}

// ---------------------------------------------------------------------------
// Weight transpose: w[K][N] fp32 -> wT[N][K] bf16 (K = N = CC)
// ---------------------------------------------------------------------------
__global__ void transpose_w(const float* __restrict__ w,
                            uint16_t* __restrict__ wT) {
    int idx = blockIdx.x * blockDim.x + threadIdx.x;   // idx = n*CC + k
    int n = idx / CC;
    int k = idx % CC;
    wT[idx] = f2bf(w[k * CC + n]);
}

// ---------------------------------------------------------------------------
// WMMA GEMM: Out[M][N] = A[M][K](bf16) * Bt[N][K](bf16)^T + bias
//   One wave computes a 32x64 tile (8 accumulators, 8 WMMA per k-step),
//   with register double-buffering of the next k-step's fragments.
//   MODE 0: bf16 row-major [M][CC]   MODE 1: bf16 Vt[b*CC+n][TT]
//   MODE 2: fp32 row-major [M][CC]
// ---------------------------------------------------------------------------
template <int MODE>
__global__ void gemm_wmma(const uint16_t* __restrict__ A,
                          const uint16_t* __restrict__ Bt,
                          const float*    __restrict__ bias,
                          void*           __restrict__ Out) {
    const int K = CC, N = CC;
    const int ntiles = N / 64;
    int mt   = blockIdx.x / ntiles;        // 32-row tile index
    int nt   = blockIdx.x % ntiles;        // 64-col tile index
    int lane = threadIdx.x & 31;
    int col  = lane & 15;
    int half = lane >> 4;

    v8f acc[2][4];
#pragma unroll
    for (int t = 0; t < 2; ++t)
#pragma unroll
        for (int j = 0; j < 4; ++j)
#pragma unroll
            for (int r = 0; r < 8; ++r) acc[t][j][r] = 0.f;

    const uint16_t* Arow0 = A + (size_t)(mt * 32 + col) * K;
    const uint16_t* Arow1 = A + (size_t)(mt * 32 + 16 + col) * K;

    FragB16 a0, a1, b[4];
    // prime k0 = 0
    a0.f4[0] = LD4(Arow0 + 8 * half);  a0.f4[1] = LD4(Arow0 + 16 + 8 * half);
    a1.f4[0] = LD4(Arow1 + 8 * half);  a1.f4[1] = LD4(Arow1 + 16 + 8 * half);
#pragma unroll
    for (int j = 0; j < 4; ++j) {
        const uint16_t* Brow = Bt + (size_t)(nt * 64 + j * 16 + col) * K + 8 * half;
        b[j].f4[0] = LD4(Brow);  b[j].f4[1] = LD4(Brow + 16);
    }

    for (int k0 = 0; k0 < K; k0 += 32) {
        int kn = (k0 + 32 < K) ? k0 + 32 : 0;   // harmless redundant prefetch at tail
        FragB16 na0, na1, nb[4];
        na0.f4[0] = LD4(Arow0 + kn + 8 * half);
        na0.f4[1] = LD4(Arow0 + kn + 16 + 8 * half);
        na1.f4[0] = LD4(Arow1 + kn + 8 * half);
        na1.f4[1] = LD4(Arow1 + kn + 16 + 8 * half);
#pragma unroll
        for (int j = 0; j < 4; ++j) {
            const uint16_t* Brow =
                Bt + (size_t)(nt * 64 + j * 16 + col) * K + kn + 8 * half;
            nb[j].f4[0] = LD4(Brow);  nb[j].f4[1] = LD4(Brow + 16);
        }
#pragma unroll
        for (int j = 0; j < 4; ++j) acc[0][j] = wmma_bf16(a0, b[j], acc[0][j]);
#pragma unroll
        for (int j = 0; j < 4; ++j) acc[1][j] = wmma_bf16(a1, b[j], acc[1][j]);
        a0 = na0; a1 = na1;
#pragma unroll
        for (int j = 0; j < 4; ++j) b[j] = nb[j];
    }

    // Epilogue. C-layout: VGPR r, lane L -> row base + r + 8*(L>=16), col L&15.
#pragma unroll
    for (int rt = 0; rt < 2; ++rt) {
        int mbase = mt * 32 + rt * 16;
#pragma unroll
        for (int j = 0; j < 4; ++j) {
            int n = nt * 64 + j * 16 + col;
            float bv = bias[n];
            if constexpr (MODE == 0) {
                uint16_t* o = (uint16_t*)Out;
#pragma unroll
                for (int r = 0; r < 8; ++r)
                    o[(size_t)(mbase + r + 8 * half) * CC + n] =
                        f2bf(acc[rt][j][r] + bv);
            } else if constexpr (MODE == 2) {
                float* o = (float*)Out;
#pragma unroll
                for (int r = 0; r < 8; ++r)
                    o[(size_t)(mbase + r + 8 * half) * CC + n] = acc[rt][j][r] + bv;
            } else {
                // Vt[b*CC + n][t]; 8 accum rows = 8 consecutive t -> one b128 store
                uint16_t* o = (uint16_t*)Out;
                int bidx = mbase / TT;
                int t    = (mbase % TT) + 8 * half;
                union { uint32_t u[4]; f32x4 f; } pk;
#pragma unroll
                for (int i = 0; i < 4; ++i) {
                    uint32_t lo = f2bf(acc[rt][j][2 * i] + bv);
                    uint32_t hi = f2bf(acc[rt][j][2 * i + 1] + bv);
                    pk.u[i] = lo | (hi << 16);
                }
                *reinterpret_cast<f32x4*>(o + (size_t)(bidx * CC + n) * TT + t) = pk.f;
            }
        }
    }
}

// ---------------------------------------------------------------------------
// Flash attention: 4 waves / block, 64 query rows per block, one (b, head).
// K/V blocks (32 keys) staged to LDS via async copies, double-buffered.
//   Q,K: bf16 [B*T][CC] row-major. Vt: bf16 [B*CC][TT]. ctx: bf16 [B*T][CC].
// ---------------------------------------------------------------------------
__global__ void __launch_bounds__(128)
attn_kernel(const uint16_t* __restrict__ Q,
            const uint16_t* __restrict__ Kb,
            const uint16_t* __restrict__ Vt,
            uint16_t*       __restrict__ ctx) {
    __shared__ __align__(16) uint16_t kbuf[2 * 32 * 64];  // [buf][key][d]   8 KB
    __shared__ __align__(16) uint16_t vbuf[2 * 64 * 32];  // [buf][d][key]   8 KB
    __shared__ __align__(16) uint16_t pbuf[4 * 16 * 32];  // per-wave P tile 4 KB

    const int tid  = threadIdx.x;
    const int w    = tid >> 5;
    const int lane = tid & 31;
    const int col  = lane & 15;
    const int half = lane >> 4;

    const int blk   = blockIdx.x;
    const int tile  = blk % (TT / 64);
    const int hn    = (blk / (TT / 64)) % NH;
    const int b     = blk / ((TT / 64) * NH);
    const int qbase = tile * 64;
    const int tq    = qbase + w * 16;          // this wave's 16 query rows

    const size_t vrow0 = (size_t)(b * NH + hn) * HD;   // Vt row base (d index)

    // stage one 32-key K/V block into LDS buffer `buf` (4 async b128 / thread)
    auto stage = [&](int buf, int s0) {
#pragma unroll
        for (int pass = 0; pass < 2; ++pass) {           // K: 32x64 bf16
            int c = tid + pass * 128;                    // chunk 0..255
            int row = c >> 3, off = (c & 7) * 8;
            async_cp16(&kbuf[buf * (32 * 64) + row * 64 + off],
                       Kb + (size_t)(b * TT + s0 + row) * CC + hn * HD + off);
        }
#pragma unroll
        for (int pass = 0; pass < 2; ++pass) {           // V: 64x32 bf16
            int c = tid + pass * 128;
            int row = c >> 2, off = (c & 3) * 8;
            async_cp16(&vbuf[buf * (64 * 32) + row * 32 + off],
                       Vt + (vrow0 + row) * TT + s0 + off);
        }
    };

    // Q A-fragments for d-windows [0,32), [32,64)
    const uint16_t* Qrow = Q + (size_t)(b * TT + tq + col) * CC + hn * HD;
    FragB16 qa[2];
#pragma unroll
    for (int f = 0; f < 2; ++f) {
        qa[f].f4[0] = LD4(Qrow + f * 32 + 16 * half);
        qa[f].f4[1] = LD4(Qrow + f * 32 + 16 * half + 8);
    }

    v8f acc[4];
#pragma unroll
    for (int j = 0; j < 4; ++j)
#pragma unroll
        for (int r = 0; r < 8; ++r) acc[j][r] = 0.f;

    float mrow[8], lrow[8];
#pragma unroll
    for (int r = 0; r < 8; ++r) { mrow[r] = -INFINITY; lrow[r] = 0.f; }

    // softmax in exp2 domain: scale' = (1/sqrt(64)) * log2(e)
    const float sc2 = 0.125f * 1.4426950408889634f;

    const int nblocks = (qbase + 64) / 32;   // keys [0, qbase+64), step 32
    stage(0, 0);
    if (nblocks > 1) stage(1, 32);

    for (int i = 0; i < nblocks; ++i) {
        const int buf = i & 1;
        const int s0  = i * 32;
        if (i + 1 < nblocks) { WAIT_ASYNCCNT(4); } else { WAIT_ASYNCCNT(0); }
        __syncthreads();                       // block `i` resident in LDS

        // ---- scores: two 16x16 tiles over keys [s0, s0+32) from LDS-K ----
        v8f s[2];
#pragma unroll
        for (int tn = 0; tn < 2; ++tn) {
#pragma unroll
            for (int r = 0; r < 8; ++r) s[tn][r] = 0.f;
            const uint16_t* kr = &kbuf[buf * (32 * 64) + (tn * 16 + col) * 64];
#pragma unroll
            for (int f = 0; f < 2; ++f) {
                FragB16 kf;   // B-frag: d = f*32 + 16h + {0..15}
                kf.f4[0] = LD4(kr + f * 32 + 16 * half);
                kf.f4[1] = LD4(kr + f * 32 + 16 * half + 8);
                s[tn] = wmma_bf16(qa[f], kf, s[tn]);
            }
        }

        // ---- scale (exp2 domain) + causal mask ----
#pragma unroll
        for (int tn = 0; tn < 2; ++tn) {
            int skey = s0 + tn * 16 + col;
#pragma unroll
            for (int r = 0; r < 8; ++r) {
                int trow = tq + r + 8 * half;
                float x  = s[tn][r] * sc2;
                s[tn][r] = (skey <= trow) ? x : -INFINITY;
            }
        }

        // ---- online softmax: DPP row reductions (rows live in 16-lane rows) ----
        float alpha[8], pv0[8], pv1[8];
#pragma unroll
        for (int r = 0; r < 8; ++r) {
            float v  = row16_max(fmaxf(s[0][r], s[1][r]));
            float nm = fmaxf(mrow[r], v);
            alpha[r] = __builtin_exp2f(mrow[r] - nm);
            mrow[r]  = nm;
            float p0 = __builtin_exp2f(s[0][r] - nm);
            float p1 = __builtin_exp2f(s[1][r] - nm);
            float rs = row16_sum(p0 + p1);
            lrow[r] = lrow[r] * alpha[r] + rs;
            pv0[r] = p0; pv1[r] = p1;
        }
#pragma unroll
        for (int j = 0; j < 4; ++j)
#pragma unroll
            for (int r = 0; r < 8; ++r) acc[j][r] *= alpha[r];

        // ---- C-layout P -> LDS -> A-layout P (wave-private; LDS is in-order
        //      within a wave, so only a compiler fence is needed) ----
        uint16_t* pw = &pbuf[w * (16 * 32)];
#pragma unroll
        for (int r = 0; r < 8; ++r) {
            pw[(r + 8 * half) * 32 + col]      = f2bf(pv0[r]);
            pw[(r + 8 * half) * 32 + 16 + col] = f2bf(pv1[r]);
        }
        asm volatile("" ::: "memory");         // stop compiler reordering st->ld
        FragB16 pa;
        pa.f4[0] = LD4(&pw[col * 32 + 16 * half]);
        pa.f4[1] = LD4(&pw[col * 32 + 16 * half + 8]);

        // ---- ctx += P * V from LDS-V (B-frag: row d, s = 16h + {0..15}) ----
#pragma unroll
        for (int j = 0; j < 4; ++j) {
            const uint16_t* vr = &vbuf[buf * (64 * 32) + (j * 16 + col) * 32 + 16 * half];
            FragB16 vf;
            vf.f4[0] = LD4(vr);  vf.f4[1] = LD4(vr + 8);
            acc[j] = wmma_bf16(pa, vf, acc[j]);
        }

        __syncthreads();                       // all waves done with buffer `buf`
        if (i + 2 < nblocks) stage(buf, (i + 2) * 32);
    }

    // ---- normalize + store ctx (bf16 row-major [B*T][CC]) ----
#pragma unroll
    for (int j = 0; j < 4; ++j)
#pragma unroll
        for (int r = 0; r < 8; ++r) {
            float o = acc[j][r] / lrow[r];
            ctx[(size_t)(b * TT + tq + r + 8 * half) * CC + hn * HD + j * 16 + col] =
                f2bf(o);
        }
}

// ---------------------------------------------------------------------------
extern "C" void kernel_launch(void* const* d_in, const int* in_sizes, int n_in,
                              void* d_out, int out_size, void* d_ws, size_t ws_size,
                              hipStream_t stream) {
    const float* src = (const float*)d_in[0];
    const float* wq  = (const float*)d_in[1];
    const float* bq  = (const float*)d_in[2];
    const float* wk  = (const float*)d_in[3];
    const float* bk  = (const float*)d_in[4];
    const float* wv  = (const float*)d_in[5];
    const float* bv  = (const float*)d_in[6];
    const float* wo  = (const float*)d_in[7];
    const float* bo  = (const float*)d_in[8];
    float*       out = (float*)d_out;

    // Workspace layout (bf16 buffers), total 48 MB.
    char* p = (char*)d_ws;
    uint16_t* srcb = (uint16_t*)p;  p += (size_t)MTOT * CC * 2;   // 8 MB
    uint16_t* wqT  = (uint16_t*)p;  p += (size_t)CC * CC * 2;     // 2 MB
    uint16_t* wkT  = (uint16_t*)p;  p += (size_t)CC * CC * 2;
    uint16_t* wvT  = (uint16_t*)p;  p += (size_t)CC * CC * 2;
    uint16_t* woT  = (uint16_t*)p;  p += (size_t)CC * CC * 2;
    uint16_t* Qb   = (uint16_t*)p;  p += (size_t)MTOT * CC * 2;   // 8 MB
    uint16_t* Kbf  = (uint16_t*)p;  p += (size_t)MTOT * CC * 2;   // 8 MB
    uint16_t* Vtb  = (uint16_t*)p;  p += (size_t)MTOT * CC * 2;   // 8 MB
    uint16_t* ctxb = (uint16_t*)p;  p += (size_t)MTOT * CC * 2;   // 8 MB

    const int nsrc = MTOT * CC;
    cvt_f32_bf16<<<(nsrc + 255) / 256, 256, 0, stream>>>(src, srcb, nsrc);
    transpose_w<<<(CC * CC) / 256, 256, 0, stream>>>(wq, wqT);
    transpose_w<<<(CC * CC) / 256, 256, 0, stream>>>(wk, wkT);
    transpose_w<<<(CC * CC) / 256, 256, 0, stream>>>(wv, wvT);
    transpose_w<<<(CC * CC) / 256, 256, 0, stream>>>(wo, woT);

    dim3 ggemm((MTOT / 32) * (CC / 64));   // 2048 waves, 32x64 tile each
    gemm_wmma<0><<<ggemm, 32, 0, stream>>>(srcb, wqT, bq, (void*)Qb);
    gemm_wmma<0><<<ggemm, 32, 0, stream>>>(srcb, wkT, bk, (void*)Kbf);
    gemm_wmma<1><<<ggemm, 32, 0, stream>>>(srcb, wvT, bv, (void*)Vtb);

    attn_kernel<<<BB * NH * (TT / 64), 128, 0, stream>>>(Qb, Kbf, Vtb, ctxb);

    gemm_wmma<2><<<ggemm, 32, 0, stream>>>(ctxb, woT, bo, (void*)out);
}
